// GATLayers_38439957299974
// MI455X (gfx1250) — matrix operand order repaired and checked by reading, not compiled, not run on hardware
//
#include <hip/hip_runtime.h>
#include <math.h>

// ---------------------------------------------------------------------------
// GAT 2-layer forward for MI455X (gfx1250, wave32).
//  - GEMM z = h @ W via v_wmma_f32_16x16x32_f16 (f16 operands, f32 accum),
//    epilogue stages C tile in LDS -> coalesced b128 stores + fused per-row
//    attention dot products (a_s, a_d) computed from the LDS tile.
//  - dest-CSR built once (histogram + scan + scatter + per-segment sort)
//  - softmax + aggregation: one wave per destination node, register-resident,
//    no atomics in the hot path; z (51 MB) is L2-resident on the 192 MB L2.
// ---------------------------------------------------------------------------

typedef __attribute__((ext_vector_type(16))) _Float16 v16h;
typedef __attribute__((ext_vector_type(8)))  _Float16 v8h;
typedef __attribute__((ext_vector_type(8)))  float    v8f;

#define D 128
#define LDS_AST 136   // padded f16 row stride for the LDS A tile (272 B)
#define ZT_ST   132   // padded f32 row stride for the LDS C tile (528 B)

// ---------------------------------------------------------------- utilities
__global__ void zero_i32_kernel(int* p, int n) {
    int i = blockIdx.x * 256 + threadIdx.x;
    if (i < n) p[i] = 0;
}

// degree histogram over E real edges + N self loops
__global__ void hist_kernel(const int* __restrict__ edge, int* __restrict__ deg,
                            int E, int N) {
    int i = blockIdx.x * 256 + threadIdx.x;
    int tot = E + N;
    if (i >= tot) return;
    int dst = (i < E) ? edge[E + i] : (i - E);
    atomicAdd(&deg[dst], 1);
}

// single-block exclusive scan; writes rowptr[0..N] and cursor copy
__global__ void scan_kernel(const int* __restrict__ deg, int* __restrict__ rowptr,
                            int* __restrict__ cursor, int N) {
    __shared__ int part[1024];
    int t = threadIdx.x;
    int chunk = (N + 1023) >> 10;
    int b = t * chunk;
    int e = b + chunk; if (e > N) e = N;
    int s = 0;
    for (int i = b; i < e; ++i) s += deg[i];
    part[t] = s;
    __syncthreads();
    if (t == 0) {
        int run = 0;
        for (int i = 0; i < 1024; ++i) { int v = part[i]; part[i] = run; run += v; }
        rowptr[N] = run;
    }
    __syncthreads();
    int run = part[t];
    for (int i = b; i < e; ++i) { rowptr[i] = run; cursor[i] = run; run += deg[i]; }
}

__global__ void scatter_kernel(const int* __restrict__ edge, int* __restrict__ cursor,
                               int* __restrict__ csr, int E, int N) {
    int i = blockIdx.x * 256 + threadIdx.x;
    int tot = E + N;
    if (i >= tot) return;
    int src, dst;
    if (i < E) { src = edge[i]; dst = edge[E + i]; }
    else       { src = i - E;  dst = i - E; }
    int pos = atomicAdd(&cursor[dst], 1);
    csr[pos] = src;
}

// make per-destination edge order deterministic (sorted by src id)
__global__ void sort_kernel(const int* __restrict__ rowptr, int* __restrict__ csr, int N) {
    int n = blockIdx.x * 256 + threadIdx.x;
    if (n >= N) return;
    int s = rowptr[n], e = rowptr[n + 1];
    for (int i = s + 1; i < e; ++i) {
        int v = csr[i]; int j = i - 1;
        while (j >= s && csr[j] > v) { csr[j + 1] = csr[j]; --j; }
        csr[j + 1] = v;
    }
}

// W[l][k][n] (f32) -> Wt[l][n][k] (f16), both layers
__global__ void convw_kernel(const float* __restrict__ W, _Float16* __restrict__ Wt) {
    int i = blockIdx.x * 256 + threadIdx.x;
    if (i >= 2 * D * D) return;
    int l = i >> 14;
    int k = (i >> 7) & 127;
    int n = i & 127;
    Wt[(size_t)l * D * D + (size_t)n * D + k] = (_Float16)W[i];
}

// ------------------------------------------------------------------- GEMM
// One block = one 16-row tile of nodes; wave w computes the 16x16 output tile
// at columns [16w, 16w+16). Epilogue: C tiles -> LDS -> coalesced stores +
// fused attention dot products.
__global__ __launch_bounds__(256) void gemm_kernel(const float* __restrict__ H,
                                                   const _Float16* __restrict__ Wt,
                                                   const float* __restrict__ att_s,
                                                   const float* __restrict__ att_d,
                                                   float* __restrict__ Z,
                                                   float* __restrict__ a_s,
                                                   float* __restrict__ a_d, int N) {
    __shared__ _Float16 Ald[16 * LDS_AST];
    __shared__ float    Zt[16 * ZT_ST];
    const int tid = threadIdx.x;
    const int rowbase = blockIdx.x * 16;

    // stage A tile (16 x 128) as f16 into LDS
    for (int i = tid; i < 16 * D; i += 256) {
        int r = i >> 7, cix = i & 127;
        int gr = rowbase + r;
        float v = (gr < N) ? H[(size_t)gr * D + cix] : 0.f;
        Ald[r * LDS_AST + cix] = (_Float16)v;
    }
    __syncthreads();

    const int wave = tid >> 5;        // output column tile 0..7
    const int lane = tid & 31;
    const int half = lane >> 4;       // 0 | 1
    const int lr   = lane & 15;       // row (A) / col (B,C)

    const _Float16* wrow = Wt + (size_t)(wave * 16 + lr) * D;

    v8f c = {};
#pragma unroll
    for (int k0 = 0; k0 < D; k0 += 32) {
        // A frag: lanes 0-15 row=lr K=k0+{0..7,16..23}; lanes 16-31 K=k0+{8..15,24..31}
        const _Float16* ap = &Ald[lr * LDS_AST + k0 + half * 8];
        v8h alo = *(const v8h*)(ap);
        v8h ahi = *(const v8h*)(ap + 16);
        v16h a = __builtin_shufflevector(alo, ahi,
                 0, 1, 2, 3, 4, 5, 6, 7, 8, 9, 10, 11, 12, 13, 14, 15);
        // B frag: lane col=lr, K = k0+{0..7,16..23} (half 0) / k0+{8..15,24..31} (half 1)
        const _Float16* bp = wrow + k0 + half * 8;
        v8h blo = *(const v8h*)(bp);
        v8h bhi = *(const v8h*)(bp + 16);
        v16h b = __builtin_shufflevector(blo, bhi,
                 0, 1, 2, 3, 4, 5, 6, 7, 8, 9, 10, 11, 12, 13, 14, 15);
        c = __builtin_amdgcn_wmma_f32_16x16x32_f16(false, a, false, b,
                                                   (short)0, c, false, false);
    }

    // C/D layout: VGPR v -> row v + half*8, col = lr; stage into LDS
#pragma unroll
    for (int v = 0; v < 8; ++v)
        Zt[(v + half * 8) * ZT_ST + wave * 16 + lr] = c[v];
    __syncthreads();

    // wave 0: fused attention dots. lane = (row, head) pair (16 rows x 2 heads)
    if (wave == 0) {
        const float* zr = &Zt[lr * ZT_ST + half * 64];
        const float* as = att_s + half * 64;
        const float* ad = att_d + half * 64;
        float ss = 0.f, sd = 0.f;
#pragma unroll
        for (int cix = 0; cix < 64; cix += 4) {
            float4 zz = *(const float4*)(zr + cix);
            float4 av = *(const float4*)(as + cix);
            float4 dv = *(const float4*)(ad + cix);
            ss += zz.x * av.x + zz.y * av.y + zz.z * av.z + zz.w * av.w;
            sd += zz.x * dv.x + zz.y * dv.y + zz.z * dv.z + zz.w * dv.w;
        }
        int gr = rowbase + lr;
        if (gr < N) { a_s[gr * 2 + half] = ss; a_d[gr * 2 + half] = sd; }
    }

    // all threads: coalesced z writeback (row per 16 threads, 2x b128 each)
    {
        int r = tid >> 4, cbase = (tid & 15) * 8;
        int gr = rowbase + r;
        if (gr < N) {
            float4 p0 = *(const float4*)(&Zt[r * ZT_ST + cbase]);
            float4 p1 = *(const float4*)(&Zt[r * ZT_ST + cbase + 4]);
            *(float4*)(Z + (size_t)gr * D + cbase)     = p0;
            *(float4*)(Z + (size_t)gr * D + cbase + 4) = p1;
        }
    }
}

// ------------------------------------- fused softmax + message aggregation
// One wave per destination node. Three register-resident passes over its edges.
__global__ __launch_bounds__(256) void aggregate_kernel(
        const float* __restrict__ Z, const float* __restrict__ a_s,
        const float* __restrict__ a_d, const int* __restrict__ rowptr,
        const int* __restrict__ csr, const float* __restrict__ bias,
        const float* __restrict__ h_prev, float* __restrict__ out,
        int N, float slope) {
    int node = blockIdx.x * 8 + (threadIdx.x >> 5);
    if (node >= N) return;
    int lane = threadIdx.x & 31;
    int start = rowptr[node], end = rowptr[node + 1];
    float2 adv = *(const float2*)(a_d + node * 2);

    // pass 1: per-head max logit
    float m0 = -INFINITY, m1 = -INFINITY;
    for (int j = start + lane; j < end; j += 32) {
        int s = csr[j];
        float2 asv = *(const float2*)(a_s + s * 2);
        float l0 = asv.x + adv.x; l0 = l0 > 0.f ? l0 : l0 * slope;
        float l1 = asv.y + adv.y; l1 = l1 > 0.f ? l1 : l1 * slope;
        m0 = fmaxf(m0, l0); m1 = fmaxf(m1, l1);
    }
#pragma unroll
    for (int off = 16; off; off >>= 1) {
        m0 = fmaxf(m0, __shfl_xor(m0, off, 32));
        m1 = fmaxf(m1, __shfl_xor(m1, off, 32));
    }

    // pass 2: per-head exp sum
    float s0 = 0.f, s1 = 0.f;
    for (int j = start + lane; j < end; j += 32) {
        int s = csr[j];
        float2 asv = *(const float2*)(a_s + s * 2);
        float l0 = asv.x + adv.x; l0 = l0 > 0.f ? l0 : l0 * slope;
        float l1 = asv.y + adv.y; l1 = l1 > 0.f ? l1 : l1 * slope;
        s0 += __expf(l0 - m0); s1 += __expf(l1 - m1);
    }
#pragma unroll
    for (int off = 16; off; off >>= 1) {
        s0 += __shfl_xor(s0, off, 32);
        s1 += __shfl_xor(s1, off, 32);
    }

    // pass 3: weighted gather of z[src]; lane owns 4 channels (head = lane>>4)
    int c0 = lane * 4;
    int head = lane >> 4;
    float mh  = head ? m1 : m0;
    float sh  = head ? s1 : s0;
    float adh = head ? adv.y : adv.x;
    float4 acc = make_float4(0.f, 0.f, 0.f, 0.f);
    for (int j = start; j < end; ++j) {
        int s = csr[j];
        float lh = a_s[s * 2 + head] + adh; lh = lh > 0.f ? lh : lh * slope;
        float alpha = __expf(lh - mh) / sh;
        const float4 zz = *(const float4*)(Z + (size_t)s * D + c0);
        acc.x += alpha * zz.x; acc.y += alpha * zz.y;
        acc.z += alpha * zz.z; acc.w += alpha * zz.w;
    }

    const float4 b4 = *(const float4*)(bias + c0);
    float4 o;
    o.x = fmaxf(acc.x + b4.x, 0.f);
    o.y = fmaxf(acc.y + b4.y, 0.f);
    o.z = fmaxf(acc.z + b4.z, 0.f);
    o.w = fmaxf(acc.w + b4.w, 0.f);
    if (h_prev) {
        const float4 hp = *(const float4*)(h_prev + (size_t)node * D + c0);
        o.x += hp.x; o.y += hp.y; o.z += hp.z; o.w += hp.w;
    }
    *(float4*)(out + (size_t)node * D + c0) = o;
}

// ---------------------------------------------------------------------------
extern "C" void kernel_launch(void* const* d_in, const int* in_sizes, int n_in,
                              void* d_out, int out_size, void* d_ws, size_t ws_size,
                              hipStream_t stream) {
    const float* x       = (const float*)d_in[0];
    const int*   edge    = (const int*)d_in[1];
    const float* W       = (const float*)d_in[2];
    const float* att_src = (const float*)d_in[3];
    const float* att_dst = (const float*)d_in[4];
    const float* bias    = (const float*)d_in[5];
    float* out = (float*)d_out;

    const int N = in_sizes[0] / D;
    const int E = in_sizes[1] / 2;
    const int Etot = E + N;

    // carve workspace (256B aligned slices)
    char* p = (char*)d_ws;
    auto carve = [&](size_t bytes) -> char* {
        char* r = p; p += (bytes + 255) & ~(size_t)255; return r;
    };
    int*      deg    = (int*)carve((size_t)N * 4);
    int*      rowptr = (int*)carve((size_t)(N + 1) * 4);
    int*      cursor = (int*)carve((size_t)N * 4);
    int*      csr    = (int*)carve((size_t)Etot * 4);
    _Float16* Wt     = (_Float16*)carve((size_t)2 * D * D * 2);
    float*    z      = (float*)carve((size_t)N * D * 4);
    float*    h1     = (float*)carve((size_t)N * D * 4);
    float*    as_    = (float*)carve((size_t)N * 2 * 4);
    float*    ad_    = (float*)carve((size_t)N * 2 * 4);

    // CSR build (graph is identical for both layers)
    zero_i32_kernel<<<(N + 255) / 256, 256, 0, stream>>>(deg, N);
    hist_kernel<<<(Etot + 255) / 256, 256, 0, stream>>>(edge, deg, E, N);
    scan_kernel<<<1, 1024, 0, stream>>>(deg, rowptr, cursor, N);
    scatter_kernel<<<(Etot + 255) / 256, 256, 0, stream>>>(edge, cursor, csr, E, N);
    sort_kernel<<<(N + 255) / 256, 256, 0, stream>>>(rowptr, csr, N);
    convw_kernel<<<(2 * D * D + 255) / 256, 256, 0, stream>>>(W, Wt);

    const int rowTiles   = (N + 15) / 16;
    const int nodeBlocks = (N + 7) / 8;
    for (int l = 0; l < 2; ++l) {
        const float* hin = (l == 0) ? x : h1;
        gemm_kernel<<<rowTiles, 256, 0, stream>>>(hin, Wt + (size_t)l * D * D,
                                                  att_src + l * D, att_dst + l * D,
                                                  z, as_, ad_, N);
        aggregate_kernel<<<nodeBlocks, 256, 0, stream>>>(
            z, as_, ad_, rowptr, csr, bias + l * D,
            (l == 0) ? nullptr : h1, (l == 0) ? h1 : out, N, 0.2f);
    }
}
